// DualAttentionBlock_10376640987855
// MI455X (gfx1250) — matrix operand order repaired and checked by reading, not compiled, not run on hardware
//
#include <hip/hip_runtime.h>
#include <hip/hip_bf16.h>
#include <cstdint>
#include <cstddef>

// DANet dual-attention block for MI455X (gfx1250), wave32 + WMMA bf16.
// B=4, C=512, Cq=32, N=4096.
// Strategy: all GEMM operands laid out so every WMMA fragment load is two
// contiguous 16B loads (global b128 from the 192MB L2, or ds_load_b128 for
// the flash probability tile). No scalar LDS gathers.

#define BD 4
#define CD 512
#define CQ 32
#define ND 4096

typedef __bf16  bf16_t;
typedef __attribute__((ext_vector_type(16))) __bf16 v16bf;
typedef __attribute__((ext_vector_type(8)))  float  v8f;

static __device__ __forceinline__ v8f zero8() {
  v8f z;
#pragma unroll
  for (int i = 0; i < 8; ++i) z[i] = 0.0f;
  return z;
}

static __device__ __forceinline__ v8f wmma_bf16(v16bf a, v16bf b, v8f c) {
  return __builtin_amdgcn_wmma_f32_16x16x32_bf16(false, a, false, b, (short)0, c,
                                                 false, false);
}

// A fragment: 16x32 bf16, row-major source (leading dim ld).
// Lane lo = row; element e -> K = (e>>3)*16 + hi*8 + (e&7): two 16B chunks.
static __device__ __forceinline__ v16bf load_a_frag(const bf16_t* p, int row0, int ld,
                                                    int k0, int lo, int hi) {
  v16bf a;
  const bf16_t* r = p + (size_t)(row0 + lo) * ld + k0;
#pragma unroll
  for (int e = 0; e < 16; ++e)
    a[e] = r[((e >> 3) << 4) + hi * 8 + (e & 7)];
  return a;
}

// B fragment: 32x16, from a matrix stored [col][k] (leading dim ldT).
// Lane lo = column; element e -> K = (e>>3)*16 + hi*8 + (e&7): two 16B chunks.
static __device__ __forceinline__ v16bf load_b_fragT(const bf16_t* pT, int col0, int ldT,
                                                     int k0, int lo, int hi) {
  v16bf b;
  const bf16_t* p = pT + (size_t)(col0 + lo) * ldT + k0;
#pragma unroll
  for (int e = 0; e < 16; ++e)
    b[e] = p[((e >> 3) << 4) + hi * 8 + (e & 7)];
  return b;
}

// ---------------------------------------------------------------- conversions
__global__ __launch_bounds__(256) void k_conv_x(const float* __restrict__ x,
                                                bf16_t* __restrict__ xbf, int n) {
  int i = blockIdx.x * 256 + threadIdx.x;
  if (i < n) xbf[i] = (bf16_t)x[i];
}

__global__ __launch_bounds__(256) void k_conv_w(const float* __restrict__ wq,
                                                const float* __restrict__ wk,
                                                const float* __restrict__ wv,
                                                bf16_t* __restrict__ wqb,
                                                bf16_t* __restrict__ wkb,
                                                bf16_t* __restrict__ wvb) {
  int i = blockIdx.x * 256 + threadIdx.x;
  if (i < CQ * CD) { wqb[i] = (bf16_t)wq[i]; wkb[i] = (bf16_t)wk[i]; }
  if (i < CD * CD) { wvb[i] = (bf16_t)wv[i]; }
}

// LDS-tiled transpose: xT[b][n][c] = bf16(x[b][c][n]); both sides coalesced.
__global__ __launch_bounds__(256) void k_make_xT(const float* __restrict__ x,
                                                 bf16_t* __restrict__ xT) {
  __shared__ bf16_t t[64 * 65];
  const int b = blockIdx.z, c0 = blockIdx.y * 64, n0 = blockIdx.x * 64;
  const int j = threadIdx.x & 63, r0 = threadIdx.x >> 6;
  const float* xb = x + ((size_t)b * CD + c0) * ND + n0;
#pragma unroll
  for (int rr = 0; rr < 16; ++rr) {
    int r = r0 + rr * 4;
    t[j * 65 + r] = (bf16_t)xb[(size_t)r * ND + j];
  }
  __syncthreads();
  bf16_t* o = xT + ((size_t)b * ND + n0) * CD + c0;
#pragma unroll
  for (int rr = 0; rr < 16; ++rr) {
    int rn = r0 + rr * 4;
    o[(size_t)rn * CD + j] = t[rn * 65 + j];
  }
}

// ------------------------------------------------------- q,k projection GEMM
// Outputs BOTH transposed: q[b][n][o], kT[b][n][o] (feeds energy WMMA directly).
__global__ __launch_bounds__(256) void k_proj_qk(const bf16_t* __restrict__ xT,
                                                 const bf16_t* __restrict__ wqb,
                                                 const bf16_t* __restrict__ wkb,
                                                 const float* __restrict__ bq,
                                                 const float* __restrict__ bk,
                                                 bf16_t* __restrict__ qb,
                                                 bf16_t* __restrict__ kT) {
  const int tid = threadIdx.x, lane = tid & 31, w = tid >> 5;
  const int lo = lane & 15, hi = lane >> 4;
  const int b = blockIdx.y, col0 = blockIdx.x * 128 + w * 16;
  const bf16_t* xTb = xT + (size_t)b * ND * CD;

  v8f acc[4];
#pragma unroll
  for (int t = 0; t < 4; ++t) acc[t] = zero8();

  for (int ks = 0; ks < CD; ks += 32) {
    v16bf bfr = load_b_fragT(xTb, col0, CD, ks, lo, hi);
    acc[0] = wmma_bf16(load_a_frag(wqb,  0, CD, ks, lo, hi), bfr, acc[0]);
    acc[1] = wmma_bf16(load_a_frag(wqb, 16, CD, ks, lo, hi), bfr, acc[1]);
    acc[2] = wmma_bf16(load_a_frag(wkb,  0, CD, ks, lo, hi), bfr, acc[2]);
    acc[3] = wmma_bf16(load_a_frag(wkb, 16, CD, ks, lo, hi), bfr, acc[3]);
  }
  const int n = col0 + lo;
#pragma unroll
  for (int r = 0; r < 8; ++r) {
    int o0 = hi * 8 + r, o1 = 16 + hi * 8 + r;
    qb[((size_t)b * ND + n) * CQ + o0] = (bf16_t)(acc[0][r] + bq[o0]);
    qb[((size_t)b * ND + n) * CQ + o1] = (bf16_t)(acc[1][r] + bq[o1]);
    kT[((size_t)b * ND + n) * CQ + o0] = (bf16_t)(acc[2][r] + bk[o0]);
    kT[((size_t)b * ND + n) * CQ + o1] = (bf16_t)(acc[3][r] + bk[o1]);
  }
}

// ------------------------------------------------------------ v projection
__global__ __launch_bounds__(256) void k_proj_v(const bf16_t* __restrict__ xT,
                                                const bf16_t* __restrict__ wvb,
                                                const float* __restrict__ bv,
                                                bf16_t* __restrict__ vb) {
  const int tid = threadIdx.x, lane = tid & 31, w = tid >> 5;
  const int lo = lane & 15, hi = lane >> 4;
  const int b = blockIdx.z, cb = blockIdx.y * 64, col0 = blockIdx.x * 128 + w * 16;
  const bf16_t* xTb = xT + (size_t)b * ND * CD;

  v8f acc[4];
#pragma unroll
  for (int t = 0; t < 4; ++t) acc[t] = zero8();

  for (int ks = 0; ks < CD; ks += 32) {
    v16bf bfr = load_b_fragT(xTb, col0, CD, ks, lo, hi);
#pragma unroll
    for (int rt = 0; rt < 4; ++rt)
      acc[rt] = wmma_bf16(load_a_frag(wvb, cb + rt * 16, CD, ks, lo, hi), bfr, acc[rt]);
  }
  const int n = col0 + lo;
#pragma unroll
  for (int rt = 0; rt < 4; ++rt)
#pragma unroll
    for (int r = 0; r < 8; ++r) {
      int c = cb + rt * 16 + hi * 8 + r;
      vb[((size_t)b * CD + c) * ND + n] = (bf16_t)(acc[rt][r] + bv[c]);
    }
}

// --------------------------------------- flash-style position attention (PA)
// One block = one batch x one 32-query tile; keys in chunks of 128.
// Energy B-frags come straight from kT (global); only ps/es live in LDS.
__global__ __launch_bounds__(256) void k_flash_pa(const bf16_t* __restrict__ qb,
                                                  const bf16_t* __restrict__ kT,
                                                  const bf16_t* __restrict__ vb,
                                                  float* __restrict__ outp) {
  __shared__ float  es[32 * 128];
  __shared__ bf16_t ps[32 * 136];
  __shared__ float  corrb[32];
  __shared__ float  rsb[32];

  const int tid = threadIdx.x, lane = tid & 31, w = tid >> 5;
  const int lo = lane & 15, hi = lane >> 4;
  const int b = blockIdx.y, nq0 = blockIdx.x * 32;

  const bf16_t* qbb = qb + (size_t)b * ND * CQ;
  const bf16_t* kTb = kT + (size_t)b * ND * CQ;
  const bf16_t* vbb = vb + (size_t)b * CD * ND;

  // Energy A-fragments: rows = queries, K = Cq = 32 (one WMMA K-step).
  v16bf qa0 = load_a_frag(qbb, nq0,      CQ, 0, lo, hi);
  v16bf qa1 = load_a_frag(qbb, nq0 + 16, CQ, 0, lo, hi);

  v8f o[4][2];   // wave owns channels [w*64, w*64+64) x 32 queries
#pragma unroll
  for (int ct = 0; ct < 4; ++ct)
#pragma unroll
    for (int nt = 0; nt < 2; ++nt) o[ct][nt] = zero8();

  // softmax state, replicated across the 8 threads serving each query row
  const int row = tid >> 3, l8 = tid & 7;
  float rm = -1e30f, rs = 0.0f;

  for (int it = 0; it < ND / 128; ++it) {
    const int m0 = it * 128;
    if (it + 1 < ND / 128) {  // global_prefetch_b8 for next chunk
      __builtin_prefetch(kTb + (size_t)(m0 + 128) * CQ, 0, 3);
      __builtin_prefetch(vbb + (size_t)(w * 64) * ND + m0 + 128, 0, 3);
    }

    // energy: wave w owns key subtile m0+16w; B-frag direct from kT
    {
      v16bf kf = load_b_fragT(kTb, m0 + w * 16, CQ, 0, lo, hi);
      v8f e0 = wmma_bf16(qa0, kf, zero8());
      v8f e1 = wmma_bf16(qa1, kf, zero8());
#pragma unroll
      for (int r = 0; r < 8; ++r) {
        es[(hi * 8 + r) * 128 + w * 16 + lo]        = e0[r];
        es[(16 + hi * 8 + r) * 128 + w * 16 + lo]   = e1[r];
      }
    }
    __syncthreads();

    // online softmax: 8 threads per row (rows stay inside one wave),
    // segment reductions via shuffles; rm/rs replicated per group.
    {
      float* er = es + row * 128 + l8 * 16;
      float cm = -1e30f;
#pragma unroll
      for (int j = 0; j < 16; ++j) cm = fmaxf(cm, er[j]);
      cm = fmaxf(cm, __shfl_xor(cm, 1, 8));
      cm = fmaxf(cm, __shfl_xor(cm, 2, 8));
      cm = fmaxf(cm, __shfl_xor(cm, 4, 8));
      float nm = fmaxf(rm, cm);
      float corr = __expf(rm - nm);
      float s = 0.0f;
      bf16_t* pr = ps + row * 136 + l8 * 16;
#pragma unroll
      for (int j = 0; j < 16; ++j) {
        float p = __expf(er[j] - nm);
        pr[j] = (bf16_t)p;
        s += p;
      }
      s += __shfl_xor(s, 1, 8);
      s += __shfl_xor(s, 2, 8);
      s += __shfl_xor(s, 4, 8);
      rs = rs * corr + s;
      rm = nm;
      if (l8 == 0) corrb[row] = corr;
    }
    __syncthreads();

    // rescale accumulators, then PV: O[c][n] += V[c][m] * P^T[m][n]
    const float c0 = corrb[lo], c1 = corrb[16 + lo];
#pragma unroll
    for (int ct = 0; ct < 4; ++ct)
#pragma unroll
      for (int nt = 0; nt < 2; ++nt) {
        float cc = nt ? c1 : c0;
#pragma unroll
        for (int r = 0; r < 8; ++r) o[ct][nt][r] *= cc;
      }
#pragma unroll
    for (int kk = 0; kk < 4; ++kk) {
      v16bf bf0 = load_b_fragT(ps, 0,  136, kk * 32, lo, hi);  // ds_load_b128
      v16bf bf1 = load_b_fragT(ps, 16, 136, kk * 32, lo, hi);
#pragma unroll
      for (int ct = 0; ct < 4; ++ct) {
        v16bf av = load_a_frag(vbb, w * 64 + ct * 16, ND, m0 + kk * 32, lo, hi);
        o[ct][0] = wmma_bf16(av, bf0, o[ct][0]);
        o[ct][1] = wmma_bf16(av, bf1, o[ct][1]);
      }
    }
  }
  if (l8 == 0) rsb[row] = rs;
  __syncthreads();

  const float i0 = 1.0f / rsb[lo], i1 = 1.0f / rsb[16 + lo];
#pragma unroll
  for (int ct = 0; ct < 4; ++ct)
#pragma unroll
    for (int nt = 0; nt < 2; ++nt) {
      float inv = nt ? i1 : i0;
#pragma unroll
      for (int r = 0; r < 8; ++r) {
        int c = w * 64 + ct * 16 + hi * 8 + r;
        int n = nq0 + nt * 16 + lo;
        outp[((size_t)b * CD + c) * ND + n] = o[ct][nt][r] * inv;
      }
    }
}

// ----------------------------------------- channel attention energy + softmax
// E[c][d] = sum_n x[c][n]*x[d][n]; softmax(rowmax-E) == exp(min-E)/sum.
// Both operands direct from row-major xbf (A: rows c; B: cols d are xbf rows).
__global__ __launch_bounds__(256) void k_ca_energy(const bf16_t* __restrict__ xbf,
                                                   bf16_t* __restrict__ attn) {
  __shared__ float es[16 * 512];

  const int tid = threadIdx.x, lane = tid & 31, w = tid >> 5;
  const int lo = lane & 15, hi = lane >> 4;
  const int b = blockIdx.y, cb = blockIdx.x * 16;
  const bf16_t* xb = xbf + (size_t)b * CD * ND;

  v8f acc[4];
#pragma unroll
  for (int t = 0; t < 4; ++t) acc[t] = zero8();

  for (int nk = 0; nk < ND; nk += 32) {
    v16bf a = load_a_frag(xb, cb, ND, nk, lo, hi);
#pragma unroll
    for (int dt = 0; dt < 4; ++dt)
      acc[dt] = wmma_bf16(a, load_b_fragT(xb, w * 64 + dt * 16, ND, nk, lo, hi), acc[dt]);
  }
#pragma unroll
  for (int dt = 0; dt < 4; ++dt)
#pragma unroll
    for (int r = 0; r < 8; ++r)
      es[(hi * 8 + r) * 512 + w * 64 + dt * 16 + lo] = acc[dt][r];
  __syncthreads();

  // softmax: 16 threads per row (within a wave), 32 d's per thread
  const int row = tid >> 4, l16 = tid & 15;
  const float* er = es + row * 512 + l16 * 32;
  float mn = 1e30f;
#pragma unroll
  for (int j = 0; j < 32; ++j) mn = fminf(mn, er[j]);
  mn = fminf(mn, __shfl_xor(mn, 1, 16));
  mn = fminf(mn, __shfl_xor(mn, 2, 16));
  mn = fminf(mn, __shfl_xor(mn, 4, 16));
  mn = fminf(mn, __shfl_xor(mn, 8, 16));
  float s = 0.0f;
#pragma unroll
  for (int j = 0; j < 32; ++j) s += __expf(mn - er[j]);
  s += __shfl_xor(s, 1, 16);
  s += __shfl_xor(s, 2, 16);
  s += __shfl_xor(s, 4, 16);
  s += __shfl_xor(s, 8, 16);
  float inv = 1.0f / s;
  bf16_t* arow = attn + ((size_t)b * CD + cb + row) * CD + l16 * 32;
#pragma unroll
  for (int j = 0; j < 32; ++j) arow[j] = (bf16_t)(__expf(mn - er[j]) * inv);
}

// --------------------- channel attention output GEMM + final combine into out
// out = gamma_pa * PA(out) + gamma_ca * (attn @ x) + 2*x
__global__ __launch_bounds__(256) void k_ca_out(const bf16_t* __restrict__ attn,
                                                const bf16_t* __restrict__ xT,
                                                const float* __restrict__ x,
                                                const float* __restrict__ gpa_p,
                                                const float* __restrict__ gca_p,
                                                float* __restrict__ out) {
  const int tid = threadIdx.x, lane = tid & 31, w = tid >> 5;
  const int lo = lane & 15, hi = lane >> 4;
  const int b = blockIdx.z, cb = blockIdx.y * 64, col0 = blockIdx.x * 128 + w * 16;
  const bf16_t* ab  = attn + (size_t)b * CD * CD;
  const bf16_t* xTb = xT + (size_t)b * ND * CD;

  v8f acc[4];
#pragma unroll
  for (int t = 0; t < 4; ++t) acc[t] = zero8();

  for (int ks = 0; ks < CD; ks += 32) {
    v16bf bfr = load_b_fragT(xTb, col0, CD, ks, lo, hi);
#pragma unroll
    for (int rt = 0; rt < 4; ++rt)
      acc[rt] = wmma_bf16(load_a_frag(ab, cb + rt * 16, CD, ks, lo, hi), bfr, acc[rt]);
  }
  const float gpa = gpa_p[0], gca = gca_p[0];
  const int n = col0 + lo;
#pragma unroll
  for (int rt = 0; rt < 4; ++rt)
#pragma unroll
    for (int r = 0; r < 8; ++r) {
      int c = cb + rt * 16 + hi * 8 + r;
      size_t idx = ((size_t)b * CD + c) * ND + n;
      out[idx] = gpa * out[idx] + gca * acc[rt][r] + 2.0f * x[idx];
    }
}

// ---------------------------------------------------------------- launcher
extern "C" void kernel_launch(void* const* d_in, const int* in_sizes, int n_in,
                              void* d_out, int out_size, void* d_ws, size_t ws_size,
                              hipStream_t stream) {
  (void)in_sizes; (void)n_in; (void)out_size; (void)ws_size;
  const float* x   = (const float*)d_in[0];
  const float* wq  = (const float*)d_in[1];
  const float* bq  = (const float*)d_in[2];
  const float* wk  = (const float*)d_in[3];
  const float* bk  = (const float*)d_in[4];
  const float* wv  = (const float*)d_in[5];
  const float* bv  = (const float*)d_in[6];
  const float* gpa = (const float*)d_in[7];
  const float* gca = (const float*)d_in[8];
  float* out = (float*)d_out;

  char* p = (char*)d_ws;
  bf16_t* xbf = (bf16_t*)p; p += (size_t)BD * CD * ND * 2;   // 16 MB, x row-major
  bf16_t* xT  = (bf16_t*)p; p += (size_t)BD * ND * CD * 2;   // 16 MB, x transposed
  bf16_t* wqb = (bf16_t*)p; p += (size_t)CQ * CD * 2;
  bf16_t* wkb = (bf16_t*)p; p += (size_t)CQ * CD * 2;
  bf16_t* wvb = (bf16_t*)p; p += (size_t)CD * CD * 2;
  bf16_t* qb  = (bf16_t*)p; p += (size_t)BD * ND * CQ * 2;
  bf16_t* kT  = (bf16_t*)p; p += (size_t)BD * ND * CQ * 2;
  bf16_t* vb  = (bf16_t*)p; p += (size_t)BD * CD * ND * 2;   // 16 MB
  bf16_t* att = (bf16_t*)p;                                  // B*512*512 bf16

  const int nx = BD * CD * ND;
  k_conv_x<<<(nx + 255) / 256, 256, 0, stream>>>(x, xbf, nx);
  k_make_xT<<<dim3(ND / 64, CD / 64, BD), 256, 0, stream>>>(x, xT);
  k_conv_w<<<(CD * CD + 255) / 256, 256, 0, stream>>>(wq, wk, wv, wqb, wkb, wvb);
  k_proj_qk<<<dim3(ND / 128, BD), 256, 0, stream>>>(xT, wqb, wkb, bq, bk, qb, kT);
  k_proj_v<<<dim3(ND / 128, CD / 64, BD), 256, 0, stream>>>(xT, wvb, bv, vb);
  k_flash_pa<<<dim3(ND / 32, BD), 256, 0, stream>>>(qb, kT, vb, out);
  k_ca_energy<<<dim3(CD / 16, BD), 256, 0, stream>>>(xbf, att);
  k_ca_out<<<dim3(ND / 128, CD / 64, BD), 256, 0, stream>>>(att, xT, x, gpa, gca, out);
}
